// GNNModel_40767829574065
// MI455X (gfx1250) — matrix operand order repaired and checked by reading, not compiled, not run on hardware
//
#include <hip/hip_runtime.h>
#include <hip/hip_bf16.h>
#include <cstddef>
#include <cstdint>

// ---------------------------------------------------------------------------
// GraphNet forward for MI455X (gfx1250).
// All GEMMs run through v_wmma_f32_16x16x32_bf16 (bf16 operands, fp32 accum).
// A-operand is staged per 16x256 K-segment in LDS (one cooperative coalesced
// global load per block, then ds_load_b128 fragment reads -> 8x less global
// A traffic than per-wave loads). Weights are pre-transposed to bf16 so each
// lane's B fragment is one contiguous 32B load. Irregular glue (embedding
// gathers, segment-sum scatters, LayerNorm, decoder reduction) are VALU
// kernels.
// ---------------------------------------------------------------------------

#define N_NODES 10000
#define N_EDGES 80000
#define N_GRAPH 64
#define DIMC    256            // D == H == 256
#define NLAYERS 4
#define LN_EPS  1e-5f

typedef __attribute__((ext_vector_type(16))) __bf16         v16bf;
typedef __attribute__((ext_vector_type(16))) unsigned short v16u;
typedef __attribute__((ext_vector_type(8)))  unsigned short v8u;
typedef __attribute__((ext_vector_type(8)))  float          v8f;

__device__ __forceinline__ unsigned short f2bf(float f) {
    union { float f; unsigned u; } x; x.f = f;
    unsigned r = x.u + 0x7FFFu + ((x.u >> 16) & 1u);   // round-to-nearest-even
    return (unsigned short)(r >> 16);
}
__device__ __forceinline__ float bf2f(unsigned short h) {
    union { unsigned u; float f; } x; x.u = ((unsigned)h) << 16;
    return x.f;
}

// ---------------------------------------------------------------------------
// WMMA GEMM:  C[M,256] = concat_seg(A)[M, 256*nseg] x Bt^T  (+bias, opt ReLU)
//   - A segments are bf16 [rows,256] with optional row indirection
//     (r = idx2[idx1[m]]), fusing the graph gathers into the GEMM.
//   - Bt is the weight matrix stored transposed+bf16: Bt[n*K + k].
//   - Block = 256 threads = 8 waves; block covers 16 rows x 256 cols.
//     Per K-segment the 16x256 A tile is staged in LDS (8 KB) once, then
//     each wave runs 8 fully-unrolled K-steps of 2 wmma (16x32 slab).
//   - M must be a multiple of 16 (true for 80000 / 10000 / 64): EXEC all-1s.
// ---------------------------------------------------------------------------
struct GemmSeg  { const unsigned short* data; const int* idx1; const int* idx2; };
struct GemmArgs {
    GemmSeg seg[4];
    int nseg;                       // K = 256 * nseg
    const unsigned short* Bt;       // bf16 [256][K]
    const float* bias;              // fp32 [256]
    unsigned short* outH;           // bf16 [M][256]
    int M;
    int relu;
};

__launch_bounds__(256)
__global__ void gemm_wmma(GemmArgs ga) {
    __shared__ unsigned short As[16 * 256];      // 8 KB A staging tile

    const int t    = threadIdx.x;
    const int lane = t & 31;
    const int wave = t >> 5;
    const int row0 = blockIdx.x * 16;
    const int hi   = lane >> 4;                  // half-wave select (ISA layout)
    const int n0   = wave * 32;                  // this wave's 32-col slab
    const int ncol = lane & 15;
    const int K    = ga.nseg * 256;

    // B fragment pointers: element e holds k = kb + 16*hi + e for col n.
    const unsigned short* __restrict__ bcol0 =
        ga.Bt + (size_t)(n0 + ncol)      * K + hi * 16;
    const unsigned short* __restrict__ bcol1 =
        ga.Bt + (size_t)(n0 + 16 + ncol) * K + hi * 16;

    // Cooperative A-staging indices: thread t loads 16 ushorts (32 B) of
    // tile row lr, columns [lc, lc+16).
    const int lr = t >> 4;
    const int lc = (t & 15) * 16;

    v8f acc0 = {}; v8f acc1 = {};

    for (int s = 0; s < ga.nseg; ++s) {
        // ---- stage A tile [16,256] of segment s into LDS (coalesced) ----
        int r = row0 + lr;
        const int* i1 = ga.seg[s].idx1;
        const int* i2 = ga.seg[s].idx2;
        if (i1) r = i1[r];
        if (i2) r = i2[r];
        const unsigned short* __restrict__ gp =
            ga.seg[s].data + (size_t)r * DIMC + lc;
        v8u w0 = *(const v8u*)(gp);
        v8u w1 = *(const v8u*)(gp + 8);
        __syncthreads();                         // previous segment fully read
        *(v8u*)(&As[lr * DIMC + lc])     = w0;
        *(v8u*)(&As[lr * DIMC + lc + 8]) = w1;
        __syncthreads();

        const int kofs = s * 256;
        #pragma unroll
        for (int kk = 0; kk < 256; kk += 32) {
            // A 16x32 bf16 fragment: lane holds row (lane&15);
            // elements 0..15 -> k = kk + 16*hi + 0..15 (matches B).
            const unsigned short* ap = &As[(lane & 15) * DIMC + kk + hi * 16];
            v8u alo = *(const v8u*)(ap);
            v8u ahi = *(const v8u*)(ap + 8);
            v16u au;
            #pragma unroll
            for (int i = 0; i < 8; ++i) { au[i] = alo[i]; au[8 + i] = ahi[i]; }
            v16bf af = __builtin_bit_cast(v16bf, au);

            v16u b0u = *(const v16u*)(bcol0 + kofs + kk);
            v16u b1u = *(const v16u*)(bcol1 + kofs + kk);
            v16bf b0 = __builtin_bit_cast(v16bf, b0u);
            v16bf b1 = __builtin_bit_cast(v16bf, b1u);

            acc0 = __builtin_amdgcn_wmma_f32_16x16x32_bf16(
                       false, af, false, b0, (short)0, acc0, false, false);
            acc1 = __builtin_amdgcn_wmma_f32_16x16x32_bf16(
                       false, af, false, b1, (short)0, acc1, false, false);
        }
    }

    // C/D layout: element r -> row = row0 + 8*hi + r, col = tile + (lane&15).
    #pragma unroll
    for (int half = 0; half < 2; ++half) {
        const v8f* acc = half ? &acc1 : &acc0;
        const int col  = n0 + half * 16 + ncol;
        const float bv = ga.bias[col];
        #pragma unroll
        for (int r = 0; r < 8; ++r) {
            const int row = row0 + hi * 8 + r;
            float v = (*acc)[r] + bv;
            if (ga.relu) v = v > 0.0f ? v : 0.0f;
            ga.outH[(size_t)row * DIMC + col] = f2bf(v);
        }
    }
}

// ---------------------------------------------------------------------------
// LayerNorm over 256 cols with optional bf16 residual delta.
// One wave per row (8 rows / 256-thread block); __shfl_xor reductions.
// Writes fp32 state in place plus a bf16 mirror for the next GEMM.
// ---------------------------------------------------------------------------
__launch_bounds__(256)
__global__ void ln_residual(const float* __restrict__ g, const float* __restrict__ beta,
                            float* __restrict__ x, const unsigned short* __restrict__ delta,
                            unsigned short* __restrict__ outH, int M) {
    const int row = blockIdx.x * 8 + (threadIdx.x >> 5);
    if (row >= M) return;
    const int lane = threadIdx.x & 31;
    float v[8];
    float s = 0.0f;
    #pragma unroll
    for (int i = 0; i < 8; ++i) {
        const int c = lane + i * 32;
        float tv = x[(size_t)row * DIMC + c];
        if (delta) tv += bf2f(delta[(size_t)row * DIMC + c]);
        v[i] = tv; s += tv;
    }
    #pragma unroll
    for (int m = 16; m >= 1; m >>= 1) s += __shfl_xor(s, m, 32);
    const float mean = s * (1.0f / 256.0f);
    float q = 0.0f;
    #pragma unroll
    for (int i = 0; i < 8; ++i) { const float d = v[i] - mean; q += d * d; }
    #pragma unroll
    for (int m = 16; m >= 1; m >>= 1) q += __shfl_xor(q, m, 32);
    const float rstd = rsqrtf(q * (1.0f / 256.0f) + LN_EPS);
    #pragma unroll
    for (int i = 0; i < 8; ++i) {
        const int c = lane + i * 32;
        const float y = (v[i] - mean) * rstd * g[c] + beta[c];
        x[(size_t)row * DIMC + c] = y;
        outH[(size_t)row * DIMC + c] = f2bf(y);
    }
}

// segment_sum scatter: out[map(r)][c] += src[r][c]  (fp32 atomics, bf16 src)
__global__ void scatter_add(const unsigned short* __restrict__ src,
                            const int* __restrict__ map1, const int* __restrict__ map2,
                            float* __restrict__ out, int rows) {
    const size_t i = (size_t)blockIdx.x * blockDim.x + threadIdx.x;
    if (i >= (size_t)rows * DIMC) return;
    const int r = (int)(i >> 8);
    const int c = (int)(i & 255);
    int tg = map1[r];
    if (map2) tg = map2[tg];
    atomicAdd(&out[(size_t)tg * DIMC + c], bf2f(src[i]));
}

__global__ void zero_f32(float* p, size_t n) {
    const size_t i = (size_t)blockIdx.x * blockDim.x + threadIdx.x;
    if (i < n) p[i] = 0.0f;
}

__global__ void cvt_bf(const float* __restrict__ in, unsigned short* __restrict__ out, size_t n) {
    const size_t i = (size_t)blockIdx.x * blockDim.x + threadIdx.x;
    if (i < n) out[i] = f2bf(in[i]);
}

// fp32 weight [K,256] -> transposed bf16 [256,K] for contiguous B fragments
__global__ void wt_cvt(const float* __restrict__ W, unsigned short* __restrict__ out, int K) {
    const size_t i = (size_t)blockIdx.x * blockDim.x + threadIdx.x;
    if (i >= (size_t)K * DIMC) return;
    const int k = (int)(i >> 8);
    const int n = (int)(i & 255);
    out[(size_t)n * K + k] = f2bf(W[i]);
}

// node[n] = concat(type_emb[x[n,0]], token_emb[x[n,1]])
__global__ void encode_nodes(const int* __restrict__ x,
                             const float* __restrict__ temb, const float* __restrict__ kemb,
                             float* __restrict__ node, unsigned short* __restrict__ nodeH) {
    const size_t i = (size_t)blockIdx.x * blockDim.x + threadIdx.x;
    if (i >= (size_t)N_NODES * DIMC) return;
    const int n = (int)(i >> 8);
    const int c = (int)(i & 255);
    const float v = (c < 128) ? temb[(size_t)x[n * 2 + 0] * 128 + c]
                              : kemb[(size_t)x[n * 2 + 1] * 128 + (c - 128)];
    node[i] = v; nodeH[i] = f2bf(v);
}

__global__ void encode_edges(const int* __restrict__ ea, const float* __restrict__ eemb,
                             float* __restrict__ edge, unsigned short* __restrict__ edgeH) {
    const size_t i = (size_t)blockIdx.x * blockDim.x + threadIdx.x;
    if (i >= (size_t)N_EDGES * DIMC) return;
    const int e = (int)(i >> 8);
    const int c = (int)(i & 255);
    const float v = eemb[(size_t)ea[e] * DIMC + c];
    edge[i] = v; edgeH[i] = f2bf(v);
}

// glob = ones(G,1) @ W[1,256] + b  ->  glob[g,c] = W[c] + b[c]
__global__ void init_glob(const float* __restrict__ Wg, const float* __restrict__ bg,
                          float* __restrict__ glob, unsigned short* __restrict__ globH) {
    const int i = blockIdx.x * blockDim.x + threadIdx.x;
    if (i >= N_GRAPH * DIMC) return;
    const int c = i & 255;
    const float v = Wg[c] + bg[c];
    glob[i] = v; globH[i] = f2bf(v);
}

// decoder l3 (256 -> 1) + mask + per-graph sum; one wave per node
__launch_bounds__(256)
__global__ void decoder_out(const unsigned short* __restrict__ h,
                            const float* __restrict__ w3, const float* __restrict__ b3,
                            const unsigned char* __restrict__ mask,
                            const int* __restrict__ batch, float* __restrict__ out) {
    const int n = blockIdx.x * 8 + (threadIdx.x >> 5);
    if (n >= N_NODES) return;
    const int lane = threadIdx.x & 31;
    float s = 0.0f;
    #pragma unroll
    for (int i = 0; i < 8; ++i) {
        const int c = lane + i * 32;
        s += bf2f(h[(size_t)n * DIMC + c]) * w3[c];
    }
    #pragma unroll
    for (int m = 16; m >= 1; m >>= 1) s += __shfl_xor(s, m, 32);
    if (lane == 0) {
        const float v = (s + b3[0]) * (mask[n] ? 1.0f : 0.0f);
        atomicAdd(&out[batch[n]], v);
    }
}

// ---------------------------------------------------------------------------
// Host side
// ---------------------------------------------------------------------------
static inline void gemm_call(hipStream_t st, int M, int relu,
                             const unsigned short* Bt, const float* bias,
                             unsigned short* outH,
                             GemmSeg s0,
                             GemmSeg s1 = {nullptr, nullptr, nullptr},
                             GemmSeg s2 = {nullptr, nullptr, nullptr},
                             GemmSeg s3 = {nullptr, nullptr, nullptr}) {
    GemmArgs ga{};
    ga.seg[0] = s0; ga.seg[1] = s1; ga.seg[2] = s2; ga.seg[3] = s3;
    ga.nseg = s3.data ? 4 : (s2.data ? 3 : (s1.data ? 2 : 1));
    ga.Bt = Bt; ga.bias = bias; ga.outH = outH;
    ga.M = M; ga.relu = relu;
    gemm_wmma<<<dim3(M / 16), dim3(256), 0, st>>>(ga);
}

extern "C" void kernel_launch(void* const* d_in, const int* in_sizes, int n_in,
                              void* d_out, int out_size, void* d_ws, size_t ws_size,
                              hipStream_t stream) {
    (void)in_sizes; (void)n_in; (void)out_size; (void)ws_size;

    // --- input layout (setup_inputs dict insertion order, recursively) ---
    // 0 x[N,2] 1 edge_index[2,E] 2 edge_attr[E,1] 3 batch[N] 4 mask[N]
    // 5 node_type_emb 6 node_token_emb 7 edge_type_emb 8 global_init.W 9 .b
    // layers l: base = 10 + 30*l:
    //   +0..5  edge_update  l1.W l1.b l2.W l2.b l3.W l3.b
    //   +6..11 node_update  ...      +12..17 global_update ...
    //   +18/19 e2m.W/b  +20/21 n2g.W/b  +22/23 e2g.W/b
    //   +24/25 edge_ln.g/b  +26/27 node_ln.g/b  +28/29 global_ln.g/b
    // 130/131 dec_ln.g/b  132..137 decoder l1.W l1.b l2.W l2.b l3.W l3.b
    const int* xin   = (const int*)d_in[0];
    const int* src   = (const int*)d_in[1];                 // edge_index[0]
    const int* dst   = (const int*)d_in[1] + N_EDGES;       // edge_index[1]
    const int* eattr = (const int*)d_in[2];
    const int* batch = (const int*)d_in[3];
    const unsigned char* mask = (const unsigned char*)d_in[4];
    auto F = [&](int i) { return (const float*)d_in[i]; };
    const int LB0 = 10;

    // --- workspace bump allocator (~246 MB total) ---
    uint8_t* wp = (uint8_t*)d_ws;
    auto alloc = [&](size_t bytes) -> void* {
        void* p = wp; wp += (bytes + 255) & ~(size_t)255; return p;
    };
    float*          nodeF = (float*)alloc((size_t)N_NODES * DIMC * 4);
    unsigned short* nodeH = (unsigned short*)alloc((size_t)N_NODES * DIMC * 2);
    float*          edgeF = (float*)alloc((size_t)N_EDGES * DIMC * 4);
    unsigned short* edgeH = (unsigned short*)alloc((size_t)N_EDGES * DIMC * 2);
    float*          globF = (float*)alloc((size_t)N_GRAPH * DIMC * 4);
    unsigned short* globH = (unsigned short*)alloc((size_t)N_GRAPH * DIMC * 2);
    float*          nmF   = (float*)alloc((size_t)N_NODES * DIMC * 4);
    unsigned short* nmH   = (unsigned short*)alloc((size_t)N_NODES * DIMC * 2);
    float*          aggNF = (float*)alloc((size_t)N_GRAPH * DIMC * 4);
    float*          aggEF = (float*)alloc((size_t)N_GRAPH * DIMC * 4);
    unsigned short* aggNH = (unsigned short*)alloc((size_t)N_GRAPH * DIMC * 2);
    unsigned short* aggEH = (unsigned short*)alloc((size_t)N_GRAPH * DIMC * 2);
    unsigned short* hA    = (unsigned short*)alloc((size_t)N_EDGES * DIMC * 2);
    unsigned short* hB    = (unsigned short*)alloc((size_t)N_EDGES * DIMC * 2);

    // transposed bf16 weights
    struct LW { unsigned short *eu1,*eu2,*eu3,*nu1,*nu2,*nu3,*gu1,*gu2,*gu3,*e2m,*n2g,*e2g; };
    LW lw[NLAYERS];
    for (int l = 0; l < NLAYERS; ++l) {
        lw[l].eu1 = (unsigned short*)alloc((size_t)1024 * DIMC * 2);
        lw[l].eu2 = (unsigned short*)alloc((size_t)256  * DIMC * 2);
        lw[l].eu3 = (unsigned short*)alloc((size_t)256  * DIMC * 2);
        lw[l].nu1 = (unsigned short*)alloc((size_t)768  * DIMC * 2);
        lw[l].nu2 = (unsigned short*)alloc((size_t)256  * DIMC * 2);
        lw[l].nu3 = (unsigned short*)alloc((size_t)256  * DIMC * 2);
        lw[l].gu1 = (unsigned short*)alloc((size_t)768  * DIMC * 2);
        lw[l].gu2 = (unsigned short*)alloc((size_t)256  * DIMC * 2);
        lw[l].gu3 = (unsigned short*)alloc((size_t)256  * DIMC * 2);
        lw[l].e2m = (unsigned short*)alloc((size_t)256  * DIMC * 2);
        lw[l].n2g = (unsigned short*)alloc((size_t)256  * DIMC * 2);
        lw[l].e2g = (unsigned short*)alloc((size_t)256  * DIMC * 2);
    }
    unsigned short* d1t = (unsigned short*)alloc((size_t)256 * DIMC * 2);
    unsigned short* d2t = (unsigned short*)alloc((size_t)256 * DIMC * 2);

    auto cvtW = [&](const float* W, unsigned short* out, int K) {
        wt_cvt<<<dim3((unsigned)(((size_t)K * DIMC + 255) / 256)), dim3(256), 0, stream>>>(W, out, K);
    };
    for (int l = 0; l < NLAYERS; ++l) {
        const int b = LB0 + 30 * l;
        cvtW(F(b + 0),  lw[l].eu1, 1024);
        cvtW(F(b + 2),  lw[l].eu2, 256);
        cvtW(F(b + 4),  lw[l].eu3, 256);
        cvtW(F(b + 6),  lw[l].nu1, 768);
        cvtW(F(b + 8),  lw[l].nu2, 256);
        cvtW(F(b + 10), lw[l].nu3, 256);
        cvtW(F(b + 12), lw[l].gu1, 768);
        cvtW(F(b + 14), lw[l].gu2, 256);
        cvtW(F(b + 16), lw[l].gu3, 256);
        cvtW(F(b + 18), lw[l].e2m, 256);
        cvtW(F(b + 20), lw[l].n2g, 256);
        cvtW(F(b + 22), lw[l].e2g, 256);
    }
    cvtW(F(132), d1t, 256);
    cvtW(F(134), d2t, 256);

    // --- encoders ---
    encode_nodes<<<dim3(N_NODES), dim3(256), 0, stream>>>(xin, F(5), F(6), nodeF, nodeH);
    encode_edges<<<dim3(N_EDGES), dim3(256), 0, stream>>>(eattr, F(7), edgeF, edgeH);
    init_glob<<<dim3((N_GRAPH * DIMC) / 256), dim3(256), 0, stream>>>(F(8), F(9), globF, globH);

    // --- message passing layers ---
    for (int l = 0; l < NLAYERS; ++l) {
        const int b = LB0 + 30 * l;
        // edge update MLP: concat(node[src], node[dst], edge, glob[batch[src]])
        gemm_call(stream, N_EDGES, 1, lw[l].eu1, F(b + 1), hA,
                  GemmSeg{nodeH, src, nullptr}, GemmSeg{nodeH, dst, nullptr},
                  GemmSeg{edgeH, nullptr, nullptr}, GemmSeg{globH, src, batch});
        gemm_call(stream, N_EDGES, 1, lw[l].eu2, F(b + 3), hB, GemmSeg{hA, nullptr, nullptr});
        gemm_call(stream, N_EDGES, 0, lw[l].eu3, F(b + 5), hA, GemmSeg{hB, nullptr, nullptr});
        ln_residual<<<dim3((N_EDGES + 7) / 8), dim3(256), 0, stream>>>(
            F(b + 24), F(b + 25), edgeF, hA, edgeH, N_EDGES);

        // messages: edge_to_message, scatter-add into node_messages
        gemm_call(stream, N_EDGES, 0, lw[l].e2m, F(b + 19), hB, GemmSeg{edgeH, nullptr, nullptr});
        zero_f32<<<dim3(N_NODES), dim3(256), 0, stream>>>(nmF, (size_t)N_NODES * DIMC);
        scatter_add<<<dim3(N_EDGES), dim3(256), 0, stream>>>(hB, dst, nullptr, nmF, N_EDGES);
        cvt_bf<<<dim3(N_NODES), dim3(256), 0, stream>>>(nmF, nmH, (size_t)N_NODES * DIMC);

        // node update MLP: concat(node, node_messages, glob[batch])
        gemm_call(stream, N_NODES, 1, lw[l].nu1, F(b + 7), hA,
                  GemmSeg{nodeH, nullptr, nullptr}, GemmSeg{nmH, nullptr, nullptr},
                  GemmSeg{globH, batch, nullptr});
        gemm_call(stream, N_NODES, 1, lw[l].nu2, F(b + 9),  hB, GemmSeg{hA, nullptr, nullptr});
        gemm_call(stream, N_NODES, 0, lw[l].nu3, F(b + 11), hA, GemmSeg{hB, nullptr, nullptr});
        ln_residual<<<dim3((N_NODES + 7) / 8), dim3(256), 0, stream>>>(
            F(b + 26), F(b + 27), nodeF, hA, nodeH, N_NODES);

        // global update: aggregate node/edge linears per graph
        gemm_call(stream, N_NODES, 0, lw[l].n2g, F(b + 21), hB, GemmSeg{nodeH, nullptr, nullptr});
        zero_f32<<<dim3((N_GRAPH * DIMC) / 256), dim3(256), 0, stream>>>(aggNF, (size_t)N_GRAPH * DIMC);
        scatter_add<<<dim3(N_NODES), dim3(256), 0, stream>>>(hB, batch, nullptr, aggNF, N_NODES);
        gemm_call(stream, N_EDGES, 0, lw[l].e2g, F(b + 23), hB, GemmSeg{edgeH, nullptr, nullptr});
        zero_f32<<<dim3((N_GRAPH * DIMC) / 256), dim3(256), 0, stream>>>(aggEF, (size_t)N_GRAPH * DIMC);
        scatter_add<<<dim3(N_EDGES), dim3(256), 0, stream>>>(hB, src, batch, aggEF, N_EDGES);
        cvt_bf<<<dim3((N_GRAPH * DIMC) / 256), dim3(256), 0, stream>>>(aggNF, aggNH, (size_t)N_GRAPH * DIMC);
        cvt_bf<<<dim3((N_GRAPH * DIMC) / 256), dim3(256), 0, stream>>>(aggEF, aggEH, (size_t)N_GRAPH * DIMC);

        gemm_call(stream, N_GRAPH, 1, lw[l].gu1, F(b + 13), hA,
                  GemmSeg{globH, nullptr, nullptr}, GemmSeg{aggNH, nullptr, nullptr},
                  GemmSeg{aggEH, nullptr, nullptr});
        gemm_call(stream, N_GRAPH, 1, lw[l].gu2, F(b + 15), hB, GemmSeg{hA, nullptr, nullptr});
        gemm_call(stream, N_GRAPH, 0, lw[l].gu3, F(b + 17), hA, GemmSeg{hB, nullptr, nullptr});
        ln_residual<<<dim3((N_GRAPH + 7) / 8), dim3(256), 0, stream>>>(
            F(b + 28), F(b + 29), globF, hA, globH, N_GRAPH);
    }

    // --- decoder: LN -> MLP(256,256) -> dot(256,1) -> mask -> segment sum ---
    ln_residual<<<dim3((N_NODES + 7) / 8), dim3(256), 0, stream>>>(
        F(130), F(131), nodeF, nullptr, nodeH, N_NODES);
    gemm_call(stream, N_NODES, 1, d1t, F(133), hA, GemmSeg{nodeH, nullptr, nullptr});
    gemm_call(stream, N_NODES, 1, d2t, F(135), hB, GemmSeg{hA, nullptr, nullptr});
    zero_f32<<<dim3(1), dim3(64), 0, stream>>>((float*)d_out, (size_t)N_GRAPH);
    decoder_out<<<dim3((N_NODES + 7) / 8), dim3(256), 0, stream>>>(
        hB, F(136), F(137), mask, batch, (float*)d_out);
}
